// MaiaEmbedder_20547123544893
// MI455X (gfx1250) — matrix operand order repaired and checked by reading, not compiled, not run on hardware
//
#include <hip/hip_runtime.h>

typedef __attribute__((ext_vector_type(16))) _Float16 v16h;
typedef __attribute__((ext_vector_type(8)))  float    v8f;
typedef __attribute__((ext_vector_type(4)))  float    v4f;
typedef unsigned long long u64;

// ---------------- geometry ----------------
// padded activation layout (f16): per channel: 10 padded rows x 16 halfs.
// padded x' in [0,9] lives at halfs 3..12 of each 16-half row (so the center
// tap window and all interior rows are 8-byte aligned); everything else is 0.
#define XROW       16
#define CIROWS     10
#define CH_STRIDE  (CIROWS * XROW)          // 160 halfs / channel
#define XBUF_BYTES (64  * CH_STRIDE * 2)    // 20480 B  (x or h buffer)
#define XIN_BYTES  (128 * CH_STRIDE * 2)    // 40960 B  (input conv, K padded to 128)
#define YBUF_BYTES (64 * 64 * 4)            // 16384 B  (f32 staging for SE)
#define SCR_FLOATS 352                      // pooled[64] se1[32] se2[128] biasA[64] biasB[64]
#define SCR_BYTES  (SCR_FLOATS * 4)
#define R1_BYTES   (XIN_BYTES + SCR_BYTES)  // 42368 B  (h | y | scratch; xin overlays h+y)
#define WAVE_LDS   (XBUF_BYTES + R1_BYTES)  // 62848 B per wave
#define WAVES_PER_BLOCK 4
#define DYN_LDS    (WAVE_LDS * WAVES_PER_BLOCK)   // 251392 B <= 320KB WGP LDS

// packed-weight buffer (f16 in d_ws):
//  conv_in : [9 taps][4 ksteps][4 mtiles][32 lanes][16 halfs] = 73728 halfs
//  12 residual convs: [9][2][4][32][16] = 36864 halfs each
#define PACK_IN_HALFS  73728
#define PACK_RES_HALFS 36864
#define PACK_TOTAL     (PACK_IN_HALFS + 12 * PACK_RES_HALFS)   // 516096 halfs

__device__ __forceinline__ void compiler_mem_barrier() { asm volatile("" ::: "memory"); }

// ---------------- weight packing ----------------
// A-fragment (16x32 f16) lane/half mapping per CDNA5 ISA:
//   row M = lane&15 ; half h -> K = (h&7) + 16*(h>>3) + 8*(lane>>4)
__global__ __launch_bounds__(256) void pack_weights(
    const float* __restrict__ w_in, const float* __restrict__ w1,
    const float* __restrict__ w2, _Float16* __restrict__ pack)
{
  const int idx = blockIdx.x * 256 + threadIdx.x;
  if (idx < PACK_IN_HALFS) {
    const int h  = idx & 15;
    const int l  = (idx >> 4) & 31;
    const int mt = (idx >> 9) & 3;
    const int ks = (idx >> 11) & 3;
    const int t  = idx >> 13;                 // 0..8
    const int co = 16 * mt + (l & 15);
    const int ci = 32 * ks + (h & 7) + 16 * ((h >> 3) & 1) + 8 * (l >> 4);
    const float v = (ci < 112) ? w_in[(co * 112 + ci) * 9 + t] : 0.0f;
    pack[idx] = (_Float16)v;
  } else if (idx < PACK_TOTAL) {
    int e = idx - PACK_IN_HALFS;
    const int c = e / PACK_RES_HALFS;         // conv 0..11 (block i = c/2; c&1 -> w2)
    e -= c * PACK_RES_HALFS;
    const int h  = e & 15;
    const int l  = (e >> 4) & 31;
    const int mt = (e >> 9) & 3;
    const int ks = (e >> 11) & 1;
    const int t  = e >> 12;                   // 0..8
    const int co = 16 * mt + (l & 15);
    const int ci = 32 * ks + (h & 7) + 16 * ((h >> 3) & 1) + 8 * (l >> 4);
    const int i  = c >> 1;
    const float* W = (c & 1) ? w2 : w1;
    pack[idx] = (_Float16)W[(((size_t)i * 64 + co) * 64 + ci) * 9 + t];
  }
}

// ---------------- helpers ----------------
__device__ __forceinline__ void zero_region(char* p, int bytes, int lane) {
  for (int o = lane * 16; o < bytes; o += 32 * 16) {
    uint4 z = {0u, 0u, 0u, 0u};
    *(uint4*)(p + o) = z;
  }
}

__device__ __forceinline__ void zacc(v8f (&acc)[4][4]) {
  const v8f z = {0.f, 0.f, 0.f, 0.f, 0.f, 0.f, 0.f, 0.f};
#pragma unroll
  for (int mt = 0; mt < 4; ++mt)
#pragma unroll
    for (int nt = 0; nt < 4; ++nt) acc[mt][nt] = z;
}

// window of 8 halfs at padded-x offset dx within a 16-half row (halfs 3+dx .. 10+dx).
// all loads are 8B aligned; shifts become immediates once dx is unrolled.
__device__ __forceinline__ void rowwin(int dx, const _Float16* rp, u64& lo, u64& hi) {
  if (dx == 1) {
    lo = *(const u64*)(rp + 4);
    hi = *(const u64*)(rp + 8);
  } else if (dx == 0) {
    const u64 a = *(const u64*)(rp + 0);
    const u64 b = *(const u64*)(rp + 4);
    const u64 c = *(const u64*)(rp + 8);
    lo = (a >> 48) | (b << 16);
    hi = (b >> 48) | (c << 16);
  } else {
    const u64 a = *(const u64*)(rp + 4);
    const u64 b = *(const u64*)(rp + 8);
    const u64 c = *(const u64*)(rp + 12);
    lo = (a >> 16) | (b << 48);
    hi = (b >> 16) | (c << 48);
  }
}

// 3x3 SAME conv, C_out=64, C_in = KS*32, over one 8x8 image resident in LDS.
// acc[mt][nt] = 16x16 f32 tile; per (tap,kstep): 4 A-frags (packed global),
// then per n-tile: build one B-frag from LDS windows and immediately issue the
// 4 mt-WMMAs (keeps only ~8 live B VGPRs -> stays off the VGPR-MSB path).
template <int KS>
__device__ __forceinline__ void conv3x3(const _Float16* __restrict__ pw,
                                        const _Float16* src, int lane,
                                        v8f (&acc)[4][4])
{
  __builtin_prefetch(pw, 0, 0);
  for (int ks = 0; ks < KS; ++ks) {
    const _Float16* srow = src + (ks * 32 + 8 * (lane >> 4)) * CH_STRIDE + ((lane & 15) >> 3) * XROW + 3 + (lane & 7);  // B lane = pixel column, half s -> ci (table order)
    for (int dy = 0; dy < 3; ++dy) {
#pragma unroll
      for (int dx = 0; dx < 3; ++dx) {
        const int t = dy * 3 + dx;
        const _Float16* pa = pw + ((size_t)(t * KS + ks) * 4) * 512 + lane * 16;
        v16h a[4];
#pragma unroll
        for (int mt = 0; mt < 4; ++mt)
          a[mt] = *(const v16h*)(pa + mt * 512);
#pragma unroll
        for (int nt = 0; nt < 4; ++nt) {
          union { v16h v; _Float16 e[16]; } bu;
#pragma unroll
          for (int s = 0; s < 16; ++s)
            bu.e[s] = srow[(2 * nt + dy) * XROW + dx + ((s & 7) + 16 * (s >> 3)) * CH_STRIDE];
#pragma unroll
          for (int mt = 0; mt < 4; ++mt)
            acc[mt][nt] = __builtin_amdgcn_wmma_f32_16x16x32_f16(
                false, a[mt], false, bu.v, (short)0, acc[mt][nt], false, false);
        }
      }
    }
  }
}

// D-tile layout: vgpr r, lane l: co = 16*mt + r + 8*(l>>4), p = 16*nt + (l&15)
template <bool RELU>
__device__ __forceinline__ void store_f16(const v8f (&acc)[4][4], _Float16* dst,
                                          const float* biasLds, int lane)
{
  const int lg = lane >> 4, ln = lane & 15;
#pragma unroll
  for (int mt = 0; mt < 4; ++mt)
#pragma unroll
    for (int r = 0; r < 8; ++r) {
      const int co = 16 * mt + r + 8 * lg;
      const float bv = biasLds[co];
#pragma unroll
      for (int nt = 0; nt < 4; ++nt) {
        const int p = 16 * nt + ln;
        float v = acc[mt][nt][r] + bv;
        if (RELU) v = fmaxf(v, 0.0f);
        dst[(co * CIROWS + (p >> 3) + 1) * XROW + 4 + (p & 7)] = (_Float16)v;
      }
    }
}

__device__ __forceinline__ void store_f32(const v8f (&acc)[4][4], float* dst,
                                          const float* biasLds, int lane)
{
  const int lg = lane >> 4, ln = lane & 15;
#pragma unroll
  for (int mt = 0; mt < 4; ++mt)
#pragma unroll
    for (int r = 0; r < 8; ++r) {
      const int co = 16 * mt + r + 8 * lg;
      const float bv = biasLds[co];
#pragma unroll
      for (int nt = 0; nt < 4; ++nt) {
        const int p = 16 * nt + ln;
        dst[co * 64 + p] = acc[mt][nt][r] + bv;
      }
    }
}

// ---------------- main fused kernel: one wave per image ----------------
__global__ __launch_bounds__(128) void maia_main(
    const float* __restrict__ input, const float* __restrict__ b_in,
    const float* __restrict__ b1, const float* __restrict__ b2,
    const float* __restrict__ sw1, const float* __restrict__ sb1,
    const float* __restrict__ sw2, const float* __restrict__ sb2,
    const _Float16* __restrict__ pack, float* __restrict__ out)
{
  extern __shared__ char smem[];
  const int lane = threadIdx.x & 31;
  const int wave = threadIdx.x >> 5;
  char* base = smem + wave * WAVE_LDS;
  _Float16* xbuf = (_Float16*)base;                 // 64ch padded f16 activations
  char* r1 = base + XBUF_BYTES;
  _Float16* hbuf = (_Float16*)r1;                   // 64ch padded f16 (conv1 out)
  float*    ybuf = (float*)(r1 + XBUF_BYTES);       // 64x64 f32 (conv2 out staging)
  float*    scr  = (float*)(r1 + XIN_BYTES);        // SE / bias scratch
  _Float16* xin  = (_Float16*)r1;                   // 128ch padded f16 (input conv only)
  const size_t img = (size_t)blockIdx.x * WAVES_PER_BLOCK + wave;

  // init: zero padded buffers (establishes zero halo rings), stage conv_in bias
  zero_region(base, XBUF_BYTES, lane);   // xbuf
  zero_region(r1, XIN_BYTES, lane);      // xin (covers hbuf+ybuf region)
  scr[224 + lane]      = b_in[lane];
  scr[224 + 32 + lane] = b_in[32 + lane];
  compiler_mem_barrier();

  // stage input image (f32 -> f16, padded layout); 2 pixels per lane per channel.
  // input is streamed exactly once across the whole dispatch -> non-temporal.
  {
    const float* ip = input + img * (112 * 64) + 2 * lane;
    const int p = 2 * lane;
    const int off0 = ((p >> 3) + 1) * XROW + 4 + (p & 7);
    for (int ch = 0; ch < 112; ++ch) {
      const float g0 = __builtin_nontemporal_load(ip + ch * 64 + 0);
      const float g1 = __builtin_nontemporal_load(ip + ch * 64 + 1);
      _Float16* d = xin + ch * CH_STRIDE + off0;
      d[0] = (_Float16)g0;
      d[1] = (_Float16)g1;
    }
  }
  compiler_mem_barrier();

  v8f acc[4][4];

  // input conv (K=112 padded to 128) + bias + relu -> xbuf
  zacc(acc);
  conv3x3<4>(pack, xin, lane, acc);
  compiler_mem_barrier();
  store_f16<true>(acc, xbuf, scr + 224, lane);
  compiler_mem_barrier();
  zero_region(r1, XBUF_BYTES, lane);     // re-establish hbuf zero halo (xin clobbered it)
  compiler_mem_barrier();

  for (int i = 0; i < 6; ++i) {
    // stage per-block conv biases into LDS scratch
    scr[224 + lane]      = b1[i * 64 + lane];
    scr[224 + 32 + lane] = b1[i * 64 + 32 + lane];
    scr[288 + lane]      = b2[i * 64 + lane];
    scr[288 + 32 + lane] = b2[i * 64 + 32 + lane];
    compiler_mem_barrier();

    // conv1 + bias + relu -> hbuf
    zacc(acc);
    conv3x3<2>(pack + PACK_IN_HALFS + (size_t)(2 * i) * PACK_RES_HALFS, xbuf, lane, acc);
    compiler_mem_barrier();
    store_f16<true>(acc, hbuf, scr + 224, lane);
    compiler_mem_barrier();

    // conv2 + bias -> ybuf (f32)
    zacc(acc);
    conv3x3<2>(pack + PACK_IN_HALFS + (size_t)(2 * i + 1) * PACK_RES_HALFS, hbuf, lane, acc);
    compiler_mem_barrier();
    store_f32(acc, ybuf, scr + 288, lane);
    compiler_mem_barrier();

    // global average pool: 2 channels per lane
#pragma unroll
    for (int hh = 0; hh < 2; ++hh) {
      const int cc = lane + 32 * hh;
      float s = 0.f;
      for (int k = 0; k < 16; ++k) {
        const float4 v = *(const float4*)(ybuf + cc * 64 + 4 * k);
        s += v.x + v.y + v.z + v.w;
      }
      scr[cc] = s * (1.0f / 64.0f);
    }
    compiler_mem_barrier();

    // SE fc1 (64 -> 32), lane = output j
    {
      float a1 = sb1[i * 32 + lane];
      for (int ch = 0; ch < 64; ++ch)
        a1 += scr[ch] * sw1[((size_t)i * 64 + ch) * 32 + lane];
      scr[64 + lane] = fmaxf(a1, 0.f);
    }
    compiler_mem_barrier();

    // SE fc2 (32 -> 128), lane owns 4 consecutive outputs
    {
      float4 a2 = *(const float4*)(sb2 + i * 128 + 4 * lane);
      for (int j = 0; j < 32; ++j) {
        const float sv = scr[64 + j];
        const float4 w = *(const float4*)(sw2 + ((size_t)i * 32 + j) * 128 + 4 * lane);
        a2.x += sv * w.x; a2.y += sv * w.y; a2.z += sv * w.z; a2.w += sv * w.w;
      }
      *(float4*)(scr + 96 + 4 * lane) = a2;
    }
    compiler_mem_barrier();

    // y = sigmoid(gate)*y + sebias ; x = relu(y + x) ; write back f16 (and out if last)
#pragma unroll
    for (int hh = 0; hh < 2; ++hh) {
      const int cc = lane + 32 * hh;
      const float g = scr[96 + cc];
      const float gate = 1.0f / (1.0f + __expf(-g));
      const float bb = scr[96 + 64 + cc];
      for (int y = 0; y < 8; ++y) {
        const float4 y0 = *(const float4*)(ybuf + cc * 64 + 8 * y);
        const float4 y1 = *(const float4*)(ybuf + cc * 64 + 8 * y + 4);
        _Float16* xr = xbuf + (cc * CIROWS + y + 1) * XROW + 4;
        float r[8];
#pragma unroll
        for (int k = 0; k < 8; ++k) {
          const float yv = (k < 4) ? ((const float*)&y0)[k] : ((const float*)&y1)[k - 4];
          const float xv = (float)xr[k];
          r[k] = fmaxf(gate * yv + bb + xv, 0.f);
        }
#pragma unroll
        for (int k = 0; k < 8; ++k) xr[k] = (_Float16)r[k];
        if (i == 5) {
          // final output is written once and never re-read -> non-temporal stores
          v4f o0 = {r[0], r[1], r[2], r[3]};
          v4f o1 = {r[4], r[5], r[6], r[7]};
          float* op = out + img * 4096 + cc * 64 + 8 * y;
          __builtin_nontemporal_store(o0, (v4f*)(op + 0));
          __builtin_nontemporal_store(o1, (v4f*)(op + 4));
        }
      }
    }
    compiler_mem_barrier();
  }
}

// ---------------- host launch ----------------
extern "C" void kernel_launch(void* const* d_in, const int* in_sizes, int n_in,
                              void* d_out, int out_size, void* d_ws, size_t ws_size,
                              hipStream_t stream) {
  (void)in_sizes; (void)n_in; (void)out_size; (void)ws_size;
  const float* input = (const float*)d_in[0];
  const float* w_in  = (const float*)d_in[1];
  const float* b_in  = (const float*)d_in[2];
  const float* w1    = (const float*)d_in[3];
  const float* b1    = (const float*)d_in[4];
  const float* w2    = (const float*)d_in[5];
  const float* b2    = (const float*)d_in[6];
  const float* sw1   = (const float*)d_in[7];
  const float* sb1   = (const float*)d_in[8];
  const float* sw2   = (const float*)d_in[9];
  const float* sb2   = (const float*)d_in[10];
  _Float16* pack = (_Float16*)d_ws;   // 516096 halfs = ~1.0 MB of workspace

  pack_weights<<<(PACK_TOTAL + 255) / 256, 256, 0, stream>>>(w_in, w1, w2, pack);

  // 8192 images, 4 per block (one wave each), 245.5KB dynamic LDS per workgroup
  maia_main<<<8192 / WAVES_PER_BLOCK, 32 * WAVES_PER_BLOCK, DYN_LDS, stream>>>(
      input, b_in, b1, b2, sw1, sb1, sw2, sb2, pack, (float*)d_out);
}
